// SAGE_2628519985358
// MI455X (gfx1250) — matrix operand order repaired and checked by reading, not compiled, not run on hardware
//
#include <hip/hip_runtime.h>

// MI455X / gfx1250 GraphSAGE link predictor.
// Strategy: bf16 activations + v_wmma_f32_16x16x32_bf16 (f32 accumulate).
// agg (100k x 256 f32 = 102.4MB) fits in 192MB L2 -> scatter atomics stay on-chip.
// Workspace usage ~258 MB (hA, hB, mean_bf, agg, deg, packed weights).

typedef __attribute__((ext_vector_type(8)))  __bf16 v8bf;
typedef __attribute__((ext_vector_type(16))) __bf16 v16bf;
typedef __attribute__((ext_vector_type(8)))  float  v8f;
typedef __attribute__((ext_vector_type(4)))  float  v4f;

#define NNODES 100000
#define DIM    256
#define NEDGE  500000
#define NPAIR  200000

// ---------------------------------------------------------------------------
// Pack a row-major [256][256] f32 weight into WMMA-B fragment order (bf16).
// Fragment (kt, nt): per ISA 7.12.2, dense 32x16 bf16 B:
//   lanes 0-15  hold K = kt*32 + {0..7, 16..23},  col = lane
//   lanes 16-31 hold K = kt*32 + {8..15, 24..31}, col = lane-16
// Stored so a lane's 16 bf16 are contiguous: idx = (((kt*16+nt)*32+lane)<<4)+e
// ---------------------------------------------------------------------------
__global__ void pack_w_bf16(const float* __restrict__ W, __bf16* __restrict__ Pk) {
    int i = blockIdx.x * blockDim.x + threadIdx.x;
    if (i >= DIM * DIM) return;
    int k = i >> 8;          // row of W (K dim)
    int n = i & 255;         // col of W (N dim)
    int kt = k >> 5, kr = k & 31;
    int nt = n >> 4, nl = n & 15;
    int lane = (((kr >> 3) & 1) << 4) | nl;
    int e = (kr & 7) | ((kr >> 4) << 3);
    Pk[(((kt * 16 + nt) * 32 + lane) << 4) + e] = (__bf16)W[k * DIM + n];
}

__global__ void f32_to_bf16_k(const float* __restrict__ src, __bf16* __restrict__ dst, long n) {
    long i = (long)blockIdx.x * blockDim.x + threadIdx.x;
    long stride = (long)gridDim.x * blockDim.x;
    for (; i < n; i += stride) dst[i] = (__bf16)src[i];
}

// b128 zero-fill (n must be a multiple of 4 floats)
__global__ void zero_f32_k(float* __restrict__ p, long n4) {
    long i = (long)blockIdx.x * blockDim.x + threadIdx.x;
    long stride = (long)gridDim.x * blockDim.x;
    v4f z = {};
    for (; i < n4; i += stride) ((v4f*)p)[i] = z;
}

// ---------------------------------------------------------------------------
// Edge scatter: one wave per edge. lane handles 8 features (one b128 bf16
// load), converts to f32, atomically accumulates into agg (L2-resident).
// Prefetch (global_prefetch_b8) pulls the source row of edge gw+64 toward L2.
// ---------------------------------------------------------------------------
__global__ void scatter_accum_k(const __bf16* __restrict__ h,
                                const int* __restrict__ esrc,
                                const int* __restrict__ edst,
                                float* __restrict__ agg,
                                float* __restrict__ deg) {
    int gw   = (blockIdx.x * blockDim.x + threadIdx.x) >> 5;
    int lane = threadIdx.x & 31;
    if (gw >= NEDGE) return;
    if (lane == 0 && gw + 64 < NEDGE) {
        __builtin_prefetch(h + (long)esrc[gw + 64] * DIM, 0, 0);
    }
    int s = esrc[gw];
    int d = edst[gw];
    v8bf m = *(const v8bf*)(h + (long)s * DIM + lane * 8);
    float* ap = agg + (long)d * DIM + lane * 8;
#pragma unroll
    for (int j = 0; j < 8; ++j) atomicAdd(ap + j, (float)m[j]);
    if (lane == 0) atomicAdd(deg + d, 1.0f);
}

__global__ void mean_bf16_k(const float* __restrict__ agg,
                            const float* __restrict__ deg,
                            __bf16* __restrict__ meanb) {
    long i = (long)blockIdx.x * blockDim.x + threadIdx.x;
    long stride = (long)gridDim.x * blockDim.x;
    long n = (long)NNODES * DIM;
    for (; i < n; i += stride) {
        float dg = deg[i >> 8];
        dg = dg < 1.0f ? 1.0f : dg;
        meanb[i] = (__bf16)(agg[i] / dg);
    }
}

// ---------------------------------------------------------------------------
// SAGE layer GEMM: h_new = act(h @ Wself + mean @ Wneigh + b)
// 128 threads = 4 waves. Block = 16 node rows x 256 cols.
// Each wave owns a 16x64 output strip (4 accumulators): A fragments are
// loaded once per k-step and reused across 4 B fragments -> 8 WMMAs / k-step.
// Epilogue stages the 16x256 bf16 tile in LDS, then writes fully coalesced
// global_store_b128s.
// ---------------------------------------------------------------------------
__global__ __launch_bounds__(128)
void sage_gemm_k(const __bf16* __restrict__ hin,
                 const __bf16* __restrict__ meanin,
                 const __bf16* __restrict__ pWs,
                 const __bf16* __restrict__ pWn,
                 const float*  __restrict__ bias,
                 int do_relu,
                 __bf16* __restrict__ hout) {
    __shared__ __bf16 ob[16 * DIM];

    const int wave  = threadIdx.x >> 5;      // col group 0..3 (64 cols each)
    const int lane  = threadIdx.x & 31;
    const int rbase = blockIdx.x * 16;
    const int arow  = rbase + (lane & 15);
    const int khalf = (lane >> 4) * 8;

    v8f acc[4] = {{}, {}, {}, {}};
#pragma unroll
    for (int kt = 0; kt < 8; ++kt) {
        const int k0 = kt * 32 + khalf;
        v8bf alo = *(const v8bf*)(hin + (long)arow * DIM + k0);
        v8bf ahi = *(const v8bf*)(hin + (long)arow * DIM + k0 + 16);
        v16bf a  = __builtin_shufflevector(alo, ahi, 0,1,2,3,4,5,6,7,8,9,10,11,12,13,14,15);
        v8bf mlo = *(const v8bf*)(meanin + (long)arow * DIM + k0);
        v8bf mhi = *(const v8bf*)(meanin + (long)arow * DIM + k0 + 16);
        v16bf am = __builtin_shufflevector(mlo, mhi, 0,1,2,3,4,5,6,7,8,9,10,11,12,13,14,15);
#pragma unroll
        for (int t = 0; t < 4; ++t) {
            const int nt = wave * 4 + t;
            v16bf b  = *(const v16bf*)(pWs + (((kt * 16 + nt) * 32 + lane) << 4));
            acc[t] = __builtin_amdgcn_wmma_f32_16x16x32_bf16(false, a,  false, b,  (short)0, acc[t], false, false);
            v16bf bn = *(const v16bf*)(pWn + (((kt * 16 + nt) * 32 + lane) << 4));
            acc[t] = __builtin_amdgcn_wmma_f32_16x16x32_bf16(false, am, false, bn, (short)0, acc[t], false, false);
        }
    }

    // C/D layout: elem r -> row r + 8*(lane>>4), col = nt*16 + (lane&15)
    const int roff = 8 * (lane >> 4);
#pragma unroll
    for (int t = 0; t < 4; ++t) {
        const int col = (wave * 4 + t) * 16 + (lane & 15);
        const float bs = bias[col];
#pragma unroll
        for (int r = 0; r < 8; ++r) {
            float v = acc[t][r] + bs;
            if (do_relu) v = v > 0.0f ? v : 0.0f;
            ob[(roff + r) * DIM + col] = (__bf16)v;
        }
    }
    __syncthreads();

    // Coalesced writeback: 128 threads x 32 bf16 (4 x b128 each)
    {
        const int base = threadIdx.x * 32;               // 128*32 = 4096 = 16*256
        const int row  = base >> 8;
        const int col  = base & 255;
        __bf16* dst = hout + (long)(rbase + row) * DIM + col;
#pragma unroll
        for (int c = 0; c < 4; ++c) {
            *(v8bf*)(dst + c * 8) = *(const v8bf*)(ob + base + c * 8);
        }
    }
}

// ---------------------------------------------------------------------------
// Fused link predictor. Block = 16 pair-rows, 512 threads (16 waves).
// z = h[s]*h[d] -> LDS; t1 = relu(z@W1+b1) -> LDS; t2 = relu(t1@W2+b2) -> LDS;
// out = t2@W3 + b3 via wave shuffle reduction.
// ---------------------------------------------------------------------------
__device__ __forceinline__
void tile_gemm_lds(const __bf16* __restrict__ A,      // LDS [16][256] bf16
                   const __bf16* __restrict__ pW,     // packed weight frags
                   const float*  __restrict__ bias,
                   __bf16* __restrict__ Out,          // LDS [16][256] bf16
                   int wave, int lane) {
    const int arow  = lane & 15;
    const int khalf = (lane >> 4) * 8;
    v8f acc = {};
#pragma unroll
    for (int kt = 0; kt < 8; ++kt) {
        const int k0 = kt * 32 + khalf;
        v8bf alo = *(const v8bf*)(A + arow * DIM + k0);
        v8bf ahi = *(const v8bf*)(A + arow * DIM + k0 + 16);
        v16bf a = __builtin_shufflevector(alo, ahi, 0,1,2,3,4,5,6,7,8,9,10,11,12,13,14,15);
        v16bf b = *(const v16bf*)(pW + (((kt * 16 + wave) * 32 + lane) << 4));
        acc = __builtin_amdgcn_wmma_f32_16x16x32_bf16(false, a, false, b, (short)0, acc, false, false);
    }
    const int col  = wave * 16 + (lane & 15);
    const float bs = bias[col];
    const int roff = 8 * (lane >> 4);
#pragma unroll
    for (int r = 0; r < 8; ++r) {
        float v = acc[r] + bs;
        v = v > 0.0f ? v : 0.0f;                       // both MLP layers use ReLU
        Out[(roff + r) * DIM + col] = (__bf16)v;
    }
}

__global__ __launch_bounds__(512)
void predictor_k(const __bf16* __restrict__ h,
                 const int* __restrict__ ps, const int* __restrict__ pd,
                 const int* __restrict__ ns, const int* __restrict__ nd,
                 const __bf16* __restrict__ pW1, const float* __restrict__ b1,
                 const __bf16* __restrict__ pW2, const float* __restrict__ b2,
                 const float*  __restrict__ W3,  const float* __restrict__ b3,
                 float* __restrict__ out) {
    __shared__ __bf16 zb[16 * DIM];
    __shared__ __bf16 t1[16 * DIM];
    __shared__ __bf16 t2[16 * DIM];

    const int tid   = threadIdx.x;
    const int wave  = tid >> 5;
    const int lane  = tid & 31;
    const int Rbase = blockIdx.x * 16;

    // Stage 1: z = h[src] * h[dst]  (wave w loads row w; lane = 8 features)
    {
        const int R = Rbase + wave;
        int s, d;
        if (R < NPAIR) { s = ps[R]; d = pd[R]; }
        else           { s = ns[R - NPAIR]; d = nd[R - NPAIR]; }
        const int c = lane * 8;
        v8bf hs = *(const v8bf*)(h + (long)s * DIM + c);
        v8bf hd = *(const v8bf*)(h + (long)d * DIM + c);
        v8bf z;
#pragma unroll
        for (int j = 0; j < 8; ++j) z[j] = (__bf16)((float)hs[j] * (float)hd[j]);
        *(v8bf*)(zb + wave * DIM + c) = z;
    }
    __syncthreads();

    tile_gemm_lds(zb, pW1, b1, t1, wave, lane);
    __syncthreads();
    tile_gemm_lds(t1, pW2, b2, t2, wave, lane);
    __syncthreads();

    // Stage 4: out[row] = t2[row] . W3 + b3   (wave w = row w)
    {
        const int c = lane * 8;
        v8bf tv = *(const v8bf*)(t2 + wave * DIM + c);
        float sum = 0.0f;
#pragma unroll
        for (int j = 0; j < 8; ++j) sum += (float)tv[j] * W3[c + j];
#pragma unroll
        for (int off = 16; off > 0; off >>= 1) sum += __shfl_xor(sum, off, 32);
        if (lane == 0) out[Rbase + wave] = sum + b3[0];
    }
}

// ---------------------------------------------------------------------------
extern "C" void kernel_launch(void* const* d_in, const int* in_sizes, int n_in,
                              void* d_out, int out_size, void* d_ws, size_t ws_size,
                              hipStream_t stream) {
    const float* x      = (const float*)d_in[0];
    const int* esrc     = (const int*)d_in[1];
    const int* edst     = (const int*)d_in[2];
    const int* pos_src  = (const int*)d_in[3];
    const int* pos_dst  = (const int*)d_in[4];
    const int* neg_src  = (const int*)d_in[5];
    const int* neg_dst  = (const int*)d_in[6];
    const float* W_self = (const float*)d_in[7];    // [3][256][256]
    const float* W_neigh= (const float*)d_in[8];    // [3][256][256]
    const float* b_sage = (const float*)d_in[9];    // [3][256]
    const float* Wp1    = (const float*)d_in[10];
    const float* bp1    = (const float*)d_in[11];
    const float* Wp2    = (const float*)d_in[12];
    const float* bp2    = (const float*)d_in[13];
    const float* Wp3    = (const float*)d_in[14];   // [256][1]
    const float* bp3    = (const float*)d_in[15];
    float* out = (float*)d_out;                     // [2P] = pos ++ neg

    // --- workspace carve-up (all offsets 256B aligned) ---
    char* ws = (char*)d_ws;
    const size_t ND_BF = (size_t)NNODES * DIM * sizeof(__bf16);   // 51.2 MB
    const size_t ND_F  = (size_t)NNODES * DIM * sizeof(float);    // 102.4 MB
    __bf16* hA      = (__bf16*)(ws);
    __bf16* hB      = (__bf16*)(ws + ND_BF);
    __bf16* mean_bf = (__bf16*)(ws + 2 * ND_BF);
    float*  agg     = (float*) (ws + 3 * ND_BF);
    float*  deg     = (float*) (ws + 3 * ND_BF + ND_F);           // contiguous after agg
    __bf16* pW      = (__bf16*)(ws + 3 * ND_BF + ND_F + ((size_t)NNODES * 4 + 255) / 256 * 256);
    const size_t WSZ = (size_t)DIM * DIM;                         // elems per packed weight
    __bf16* pWs  = pW;                  // 3 slots
    __bf16* pWn  = pW + 3 * WSZ;        // 3 slots
    __bf16* pWp1 = pW + 6 * WSZ;
    __bf16* pWp2 = pW + 7 * WSZ;

    // --- pack weights (8 launches, 65536 threads each) ---
    const int packGrid = (DIM * DIM + 255) / 256;
    for (int l = 0; l < 3; ++l) {
        pack_w_bf16<<<packGrid, 256, 0, stream>>>(W_self  + (size_t)l * WSZ, pWs + (size_t)l * WSZ);
        pack_w_bf16<<<packGrid, 256, 0, stream>>>(W_neigh + (size_t)l * WSZ, pWn + (size_t)l * WSZ);
    }
    pack_w_bf16<<<packGrid, 256, 0, stream>>>(Wp1, pWp1);
    pack_w_bf16<<<packGrid, 256, 0, stream>>>(Wp2, pWp2);

    // --- x -> bf16 ---
    f32_to_bf16_k<<<4096, 256, 0, stream>>>(x, hA, (long)NNODES * DIM);

    // --- 3 SAGE layers ---
    __bf16* hcur = hA;
    __bf16* hnxt = hB;
    for (int l = 0; l < 3; ++l) {
        zero_f32_k<<<4096, 256, 0, stream>>>(agg, ((long)NNODES * DIM + NNODES) / 4);
        scatter_accum_k<<<NEDGE / 8, 256, 0, stream>>>(hcur,
                                                       esrc + (size_t)l * NEDGE,
                                                       edst + (size_t)l * NEDGE,
                                                       agg, deg);
        mean_bf16_k<<<4096, 256, 0, stream>>>(agg, deg, mean_bf);
        sage_gemm_k<<<NNODES / 16, 128, 0, stream>>>(hcur, mean_bf,
                                                     pWs + (size_t)l * WSZ,
                                                     pWn + (size_t)l * WSZ,
                                                     b_sage + (size_t)l * DIM,
                                                     (l < 2) ? 1 : 0,
                                                     hnxt);
        __bf16* t = hcur; hcur = hnxt; hnxt = t;
    }
    // after 3 swaps, final h is in hcur (== hB)

    // --- fused predictor over 2P rows ---
    predictor_k<<<(2 * NPAIR) / 16, 512, 0, stream>>>(hcur,
                                                      pos_src, pos_dst, neg_src, neg_dst,
                                                      pWp1, bp1, pWp2, bp2, Wp3, bp3,
                                                      out);
}
